// Int8GPTJAttention_12395275616370
// MI455X (gfx1250) — 8-bit matrix operand order repaired and checked by reading, compiled, not run on hardware
//
#include <hip/hip_runtime.h>
#include <hip/hip_bf16.h>
#include <stdint.h>

#define B_   2
#define S_   1024
#define E_   4096
#define H_   16
#define D_   256
#define RD_  64
#define MS_  (B_ * S_)
#define QBLK 32
#define A_QKV_ 0.00012f
#define A_QK_  0.0002f
#define A_PV_  (1.0f / 127.0f)
#define A_O_   0.01f
#define NEG_HUGE (-3.402823466e38f)

#define BLD  80
#define SCLD 1032
#define PLD  1032

typedef int v8i __attribute__((ext_vector_type(8)));

template <bool SA, bool SB>
__device__ __forceinline__ v8i wmma8(v8i a, v8i b, v8i c) {
  return __builtin_amdgcn_wmma_i32_16x16x64_iu8(SA, a, SB, b, c, false, false);
}

__device__ __forceinline__ v8i load_a_i8(const int8_t* base, int lda) {
  const int lane = threadIdx.x & 31;
  const int m    = lane & 15;
  const int half = lane >> 4;
  const int8_t* r = base + m * lda + 16 * half;
  v8i f;
#pragma unroll
  for (int p = 0; p < 4; ++p) {
    int2 d = *(const int2*)(r + 8 * p + 16 * (p >> 1));
    f[2 * p]     = d.x;
    f[2 * p + 1] = d.y;
  }
  return f;
}

__device__ __forceinline__ v8i load_b_i8(const int8_t* base, int ldb) {
  const int lane = threadIdx.x & 31;
  const int n    = lane & 15;
  const int half = lane >> 4;
  const int8_t* c = base + n * ldb + 16 * half;
  int4 lo = *(const int4*)(c);
  int4 hi = *(const int4*)(c + 32);
  v8i f;
  f[0] = lo.x; f[1] = lo.y; f[2] = lo.z; f[3] = lo.w;
  f[4] = hi.x; f[5] = hi.y; f[6] = hi.z; f[7] = hi.w;
  return f;
}

__global__ void quant_kernel(const float* __restrict__ in,
                             int8_t* __restrict__ out, int n) {
  for (int i = blockIdx.x * blockDim.x + threadIdx.x; i < n;
       i += gridDim.x * blockDim.x) {
    float v = fminf(fmaxf(rintf(in[i]), -128.f), 127.f);
    out[i] = (int8_t)v;
  }
}

__global__ void gemm_i8_kernel(const int8_t* __restrict__ A,
                               const int8_t* __restrict__ W,
                               const float* __restrict__ bias, float alpha,
                               int M, int N, int K, int mode,
                               int8_t* __restrict__ out8,
                               float* __restrict__ outf) {
  __shared__ int8_t bsmem[2][64 * BLD];

  const int wave = threadIdx.x >> 5;
  const int lane = threadIdx.x & 31;
  const int half = lane >> 4;
  const int ncol = lane & 15;
  const int m0 = blockIdx.y * 128 + wave * 16;
  const int n0 = blockIdx.x * 64;
  const int nk = K / 64;

  const int tn   = threadIdx.x >> 2;
  const int tseg = (threadIdx.x & 3) * 16;

  auto issue_tile = [&](int kt, int buf) {
    const uint64_t ga =
        (uint64_t)(uintptr_t)(W + (size_t)(n0 + tn) * K + (size_t)kt * 64 + tseg);
    const uint32_t la = (uint32_t)(uintptr_t)&bsmem[buf][tn * BLD + tseg];
    asm volatile("global_load_async_to_lds_b128 %0, %1, off"
                 :: "v"(la), "v"(ga) : "memory");
  };

  v8i acc0 = {0,0,0,0,0,0,0,0};
  v8i acc1 = {0,0,0,0,0,0,0,0};
  v8i acc2 = {0,0,0,0,0,0,0,0};
  v8i acc3 = {0,0,0,0,0,0,0,0};

  issue_tile(0, 0);
  if (nk > 1) issue_tile(1, 1);

  const int8_t* arow = A + (size_t)m0 * K;
  for (int kt = 0; kt < nk; ++kt) {
    const int buf = kt & 1;
    if (kt + 1 < nk) asm volatile("s_wait_asynccnt 0x1" ::: "memory");
    else             asm volatile("s_wait_asynccnt 0x0" ::: "memory");
    __syncthreads();

    v8i a  = load_a_i8(arow + kt * 64, K);
    v8i b0 = load_b_i8(&bsmem[buf][ 0 * BLD], BLD);
    v8i b1 = load_b_i8(&bsmem[buf][16 * BLD], BLD);
    v8i b2 = load_b_i8(&bsmem[buf][32 * BLD], BLD);
    v8i b3 = load_b_i8(&bsmem[buf][48 * BLD], BLD);
    acc0 = wmma8<true, true>(a, b0, acc0);
    acc1 = wmma8<true, true>(a, b1, acc1);
    acc2 = wmma8<true, true>(a, b2, acc2);
    acc3 = wmma8<true, true>(a, b3, acc3);

    __syncthreads();
    if (kt + 2 < nk) issue_tile(kt + 2, buf);
  }

  v8i accs[4] = {acc0, acc1, acc2, acc3};
#pragma unroll
  for (int j = 0; j < 4; ++j) {
#pragma unroll
    for (int r = 0; r < 8; ++r) {
      const int m = m0 + r + 8 * half;
      const int n = n0 + 16 * j + ncol;
      const float y = alpha * (float)accs[j][r] + bias[n];
      if (mode == 3) {
        outf[(size_t)m * N + n] = y;
      } else {
        const float q = fminf(fmaxf(rintf(y), -128.f), 127.f);
        const int bb = m >> 10, s = m & (S_ - 1);
        const int h = n >> 8, d = n & (D_ - 1);
        if (mode == 2)
          out8[(((size_t)(bb * H_ + h)) * D_ + d) * S_ + s] = (int8_t)q;
        else
          out8[(((size_t)(bb * H_ + h)) * S_ + s) * D_ + d] = (int8_t)q;
      }
    }
  }
}

__global__ void rope_kernel(int8_t* __restrict__ q8, int8_t* __restrict__ k8) {
  const int total = B_ * H_ * S_ * (RD_ / 2);
  const int idx = blockIdx.x * blockDim.x + threadIdx.x;
  if (idx >= total) return;
  const int j  = idx & (RD_ / 2 - 1);
  const int s  = (idx >> 5) & (S_ - 1);
  const int bh = idx >> 15;
  const float inv_freq = powf(10000.f, -(float)(2 * j) / (float)RD_);
  const float ang = (float)s * inv_freq;
  const float sn = sinf(ang), cs = cosf(ang);
  const size_t base = ((size_t)bh * S_ + s) * D_ + 2 * j;
  int8_t* ptrs[2] = {q8, k8};
#pragma unroll
  for (int t = 0; t < 2; ++t) {
    const float x1 = (float)ptrs[t][base];
    const float x2 = (float)ptrs[t][base + 1];
    float r1 = x1 * cs - x2 * sn;
    float r2 = x2 * cs + x1 * sn;
    r1 = truncf(fminf(fmaxf(r1, -128.f), 127.f));
    r2 = truncf(fminf(fmaxf(r2, -128.f), 127.f));
    ptrs[t][base]     = (int8_t)r1;
    ptrs[t][base + 1] = (int8_t)r2;
  }
}

__global__ void attn_kernel(const int8_t* __restrict__ q8,
                            const int8_t* __restrict__ k8,
                            const int8_t* __restrict__ vT,
                            int8_t* __restrict__ o8) {
  extern __shared__ char smem[];
  float*  sc = (float*)smem;
  int8_t* pl = (int8_t*)(smem + (size_t)QBLK * SCLD * sizeof(float));

  const int bh   = blockIdx.y;
  const int q0   = blockIdx.x * QBLK;
  const int wave = threadIdx.x >> 5;
  const int lane = threadIdx.x & 31;
  const int half = lane >> 4;
  const int ncol = lane & 15;

  const int8_t* qb  = q8 + (size_t)bh * S_ * D_ + (size_t)q0 * D_;
  const int8_t* kbp = k8 + (size_t)bh * S_ * D_;
  const float sscale = A_QK_ / 16.0f;

  for (int mt = 0; mt < QBLK / 16; ++mt) {
    v8i af0 = load_a_i8(qb + mt * 16 * D_ +   0, D_);
    v8i af1 = load_a_i8(qb + mt * 16 * D_ +  64, D_);
    v8i af2 = load_a_i8(qb + mt * 16 * D_ + 128, D_);
    v8i af3 = load_a_i8(qb + mt * 16 * D_ + 192, D_);
    for (int nt = wave; nt < S_ / 16; nt += 8) {
      const int key = nt * 16 + ncol;
      if (nt * 16 > q0 + QBLK - 1) {
#pragma unroll
        for (int r = 0; r < 8; ++r)
          sc[(mt * 16 + r + 8 * half) * SCLD + key] = NEG_HUGE;
        continue;
      }
      v8i acc = {0,0,0,0,0,0,0,0};
      acc = wmma8<true, true>(af0, load_b_i8(kbp + (size_t)nt * 16 * D_ +   0, D_), acc);
      acc = wmma8<true, true>(af1, load_b_i8(kbp + (size_t)nt * 16 * D_ +  64, D_), acc);
      acc = wmma8<true, true>(af2, load_b_i8(kbp + (size_t)nt * 16 * D_ + 128, D_), acc);
      acc = wmma8<true, true>(af3, load_b_i8(kbp + (size_t)nt * 16 * D_ + 192, D_), acc);
#pragma unroll
      for (int r = 0; r < 8; ++r) {
        const int qrow = q0 + mt * 16 + r + 8 * half;
        const float v = (key <= qrow) ? (float)acc[r] * sscale : NEG_HUGE;
        sc[(mt * 16 + r + 8 * half) * SCLD + key] = v;
      }
    }
  }
  __syncthreads();

  {
    const int row = threadIdx.x >> 3;
    const int sub = threadIdx.x & 7;
    float mx = NEG_HUGE;
    for (int c = sub; c < S_; c += 8) mx = fmaxf(mx, sc[row * SCLD + c]);
    for (int o = 1; o < 8; o <<= 1) mx = fmaxf(mx, __shfl_xor(mx, o, 32));
    float sm = 0.f;
    for (int c = sub; c < S_; c += 8) {
      const float e = expf(sc[row * SCLD + c] - mx);
      sc[row * SCLD + c] = e;
      sm += e;
    }
    for (int o = 1; o < 8; o <<= 1) sm += __shfl_xor(sm, o, 32);
    const float inv = 127.f / sm;
    for (int c = sub; c < S_; c += 8)
      pl[row * PLD + c] = (int8_t)(int)rintf(sc[row * SCLD + c] * inv);
  }
  __syncthreads();

  const int8_t* vb  = vT + (size_t)bh * D_ * S_;
  const int nkb = (q0 + QBLK + 63) >> 6;
  for (int mt = 0; mt < QBLK / 16; ++mt) {
    for (int nt = wave; nt < D_ / 16; nt += 8) {
      v8i acc = {0,0,0,0,0,0,0,0};
      for (int kb = 0; kb < nkb; ++kb) {
        v8i a = load_a_i8(pl + mt * 16 * PLD + kb * 64, PLD);
        v8i b = load_b_i8(vb + (size_t)nt * 16 * S_ + kb * 64, S_);
        acc = wmma8<false, true>(a, b, acc);
      }
#pragma unroll
      for (int r = 0; r < 8; ++r) {
        const int qrow = q0 + mt * 16 + r + 8 * half;
        const int d = nt * 16 + ncol;
        const float o = fminf(fmaxf(rintf((float)acc[r] * A_PV_), -128.f), 127.f);
        const int bb = bh >> 4, h = bh & (H_ - 1);
        o8[((size_t)(bb * S_ + qrow)) * E_ + h * D_ + d] = (int8_t)o;
      }
    }
  }
}

extern "C" void kernel_launch(void* const* d_in, const int* in_sizes, int n_in,
                              void* d_out, int out_size, void* d_ws,
                              size_t ws_size, hipStream_t stream) {
  const float* hs = (const float*)d_in[0];
  const float* Wq = (const float*)d_in[1];
  const float* Wk = (const float*)d_in[2];
  const float* Wv = (const float*)d_in[3];
  const float* Wo = (const float*)d_in[4];
  const float* bq = (const float*)d_in[5];
  const float* bk = (const float*)d_in[6];
  const float* bv = (const float*)d_in[7];
  const float* bo = (const float*)d_in[8];

  const size_t XSZ = (size_t)MS_ * E_;
  const size_t WSZ = (size_t)E_ * E_;
  int8_t* x8  = (int8_t*)d_ws;
  int8_t* q8  = x8 + XSZ;
  int8_t* k8  = q8 + XSZ;
  int8_t* vT  = k8 + XSZ;
  int8_t* o8  = vT + XSZ;
  int8_t* Wq8 = o8 + XSZ;
  int8_t* Wk8 = Wq8 + WSZ;
  int8_t* Wv8 = Wk8 + WSZ;
  int8_t* Wo8 = Wv8 + WSZ;

  quant_kernel<<<2048, 256, 0, stream>>>(hs, x8, (int)XSZ);
  quant_kernel<<<2048, 256, 0, stream>>>(Wq, Wq8, (int)WSZ);
  quant_kernel<<<2048, 256, 0, stream>>>(Wk, Wk8, (int)WSZ);
  quant_kernel<<<2048, 256, 0, stream>>>(Wv, Wv8, (int)WSZ);
  quant_kernel<<<2048, 256, 0, stream>>>(Wo, Wo8, (int)WSZ);

  dim3 gg(E_ / 64, MS_ / 128);
  gemm_i8_kernel<<<gg, 256, 0, stream>>>(x8, Wq8, bq, A_QKV_, MS_, E_, E_, 0, q8, nullptr);
  gemm_i8_kernel<<<gg, 256, 0, stream>>>(x8, Wk8, bk, A_QKV_, MS_, E_, E_, 1, k8, nullptr);
  gemm_i8_kernel<<<gg, 256, 0, stream>>>(x8, Wv8, bv, A_QKV_, MS_, E_, E_, 2, vT, nullptr);

  rope_kernel<<<(B_ * H_ * S_ * (RD_ / 2)) / 256, 256, 0, stream>>>(q8, k8);

  const size_t smem =
      (size_t)QBLK * SCLD * sizeof(float) + (size_t)QBLK * PLD;
  attn_kernel<<<dim3(S_ / QBLK, B_ * H_), 256, smem, stream>>>(q8, k8, vT, o8);

  gemm_i8_kernel<<<gg, 256, 0, stream>>>(o8, Wo8, bo, A_O_, MS_, E_, E_, 3,
                                         nullptr, (float*)d_out);
}
